// RGCN_82746839924852
// MI455X (gfx1250) — matrix operand order repaired and checked by reading, not compiled, not run on hardware
//
#include <hip/hip_runtime.h>
#include <hip/hip_bf16.h>

typedef __attribute__((ext_vector_type(16))) __bf16 v16bf;
typedef __attribute__((ext_vector_type(8)))  float  v8f;

union Frag16 { v16bf v; uint4 u[2]; };

#define RELS 4
#define OUTC 256

// ---------------------------------------------------------------------------
// generic u32 fill (used for zeroing / init of scratch every call)
// ---------------------------------------------------------------------------
__global__ void rgcn_fill_u32(unsigned* p, unsigned v, size_t n) {
    size_t i = (size_t)blockIdx.x * blockDim.x + threadIdx.x;
    size_t stride = (size_t)gridDim.x * blockDim.x;
    for (; i < n; i += stride) p[i] = v;
}

// ---------------------------------------------------------------------------
// pack f32 weight [K,256] row-major -> bf16 WMMA B-fragment layout
// dst[((kt*16+nt)*32 + lane)*16 + idx] = W[k*256 + n]
//   n = nt*16 + (lane&15),  k = kt*32 + (lane<16 ? 0 : 16) + idx
// ---------------------------------------------------------------------------
__global__ void rgcn_pack_w(const float* __restrict__ W, __bf16* __restrict__ dst, int K) {
    int t = blockIdx.x * 256 + threadIdx.x;
    if (t >= K * OUTC) return;
    int idx  = t & 15;
    int lane = (t >> 4) & 31;
    int nt   = (t >> 9) & 15;
    int kt   = t >> 13;
    int n = nt * 16 + (lane & 15);
    int k = kt * 32 + ((lane < 16) ? 0 : 16) + idx;
    dst[t] = (__bf16)W[k * OUTC + n];
}

// ---------------------------------------------------------------------------
// embedding gather -> bf16 feats [Npad, 384], pad rows zeroed
// ---------------------------------------------------------------------------
__global__ void rgcn_gather(const int* __restrict__ x,
                            const float* __restrict__ e0, const float* __restrict__ e1,
                            const float* __restrict__ e2, const float* __restrict__ e3,
                            const float* __restrict__ e4, const float* __restrict__ e5,
                            __bf16* __restrict__ feats, int Nn, int Npad) {
    size_t t = (size_t)blockIdx.x * 256 + threadIdx.x;
    size_t total = (size_t)Npad * 384;
    if (t >= total) return;
    int n = (int)(t / 384);
    int c = (int)(t % 384);
    float v = 0.f;
    if (n < Nn) {
        const float* tabs[6] = {e0, e1, e2, e3, e4, e5};
        int seg = c >> 6, cc = c & 63;
        int id = x[n * 6 + seg];
        v = tabs[seg][id * 64 + cc];
    }
    feats[t] = (__bf16)v;
}

// ---------------------------------------------------------------------------
// bf16 WMMA GEMM, fully software-pipelined:
//  - wave computes a 16x128 tile: 8 WMMAs per k-step
//  - block (8 waves) cooperatively double-buffers the 8KB B panel in LDS;
//    next k-step's global loads overlap current k-step's WMMAs (loadcnt)
//  - all 8 B fragments are loaded into distinct register sets BEFORE the
//    WMMA burst: 16 back-to-back ds_load_b128 amortize one LDS round-trip,
//    then 8 WMMAs issue with a single satisfied dscnt wait (no per-frag
//    WAR-forced ds stalls)
//  grid = (Npad/128, 2, nmat), block = 256
// ---------------------------------------------------------------------------
__global__ void rgcn_wmma_gemm(const __bf16* __restrict__ A,
                               const __bf16* __restrict__ Bp,
                               float* __restrict__ C,
                               int K, size_t bStride, size_t cStride) {
    __shared__ __bf16 Bsh[2][4096];   // 2 x 8KB double buffer (8 n-tiles)

    const int tid  = threadIdx.x;
    const int wave = tid >> 5;
    const int lane = tid & 31;
    const int sub  = (lane >> 4) & 1;
    const int l16  = lane & 15;

    const int m0 = blockIdx.x * 128 + wave * 16;
    const int q  = blockIdx.y;                       // 128-col group 0..1
    const __bf16* B = Bp + (size_t)blockIdx.z * bStride;
    float* Cm = C + (size_t)blockIdx.z * cStride;
    const int ksteps = K >> 5;

    // cooperative B-stage mapping: thread -> (tile j, 32B chunk r16)
    const int j   = tid >> 5;                        // 0..7
    const int r16 = (tid & 31) * 16;                 // elem offset in tile
    const __bf16* bsrc = B + ((size_t)(q * 8 + j) << 9) + r16;
    const size_t  bAdv = (size_t)16 * 512;           // elems per k-step

    const __bf16* arow = A + (size_t)(m0 + l16) * K + sub * 16;

    // ---- prologue: stage k-step 0 ----
    {
        const uint4* s = (const uint4*)bsrc;
        uint4 b0 = s[0], b1 = s[1];
        uint4* d = (uint4*)(&Bsh[0][j * 512 + r16]);
        d[0] = b0;
        d[1] = b1;
    }
    Frag16 a_cur;
    a_cur.u[0] = ((const uint4*)arow)[0];
    a_cur.u[1] = ((const uint4*)arow)[1];
    __syncthreads();

    v8f acc[8];
    #pragma unroll
    for (int t = 0; t < 8; ++t)
        #pragma unroll
        for (int i = 0; i < 8; ++i) acc[t][i] = 0.f;

    for (int kt = 0; kt < ksteps; ++kt) {
        const int cur = kt & 1;
        const bool more = (kt + 1 < ksteps);

        // issue next k-step's global loads (overlap with this step's WMMAs)
        uint4 bn0, bn1;
        Frag16 a_nxt;
        if (more) {
            const uint4* s = (const uint4*)(bsrc + (size_t)(kt + 1) * bAdv);
            bn0 = s[0];
            bn1 = s[1];
            const __bf16* an = arow + ((kt + 1) << 5);
            a_nxt.u[0] = ((const uint4*)an)[0];
            a_nxt.u[1] = ((const uint4*)an)[1];
        }

        // batch all 16 LDS fragment loads (distinct dest regs -> no WAR chain)
        Frag16 bfr[8];
        #pragma unroll
        for (int t = 0; t < 8; ++t) {
            const uint4* bp = (const uint4*)(&Bsh[cur][t * 512 + lane * 16]);
            bfr[t].u[0] = bp[0];
            bfr[t].u[1] = bp[1];
        }
        // WMMA burst
        #pragma unroll
        for (int t = 0; t < 8; ++t) {
            acc[t] = __builtin_amdgcn_wmma_f32_16x16x32_bf16(
                false, a_cur.v, false, bfr[t].v, (short)0, acc[t], false, false);
        }

        // stage next panel into the alternate buffer
        if (more) {
            uint4* d = (uint4*)(&Bsh[cur ^ 1][j * 512 + r16]);
            d[0] = bn0;
            d[1] = bn1;
            a_cur = a_nxt;
        }
        __syncthreads();
    }

    // store: VGPR v -> row m0 + sub*8 + v, col = q*128 + t*16 + l16
    #pragma unroll
    for (int t = 0; t < 8; ++t) {
        int ncol = q * 128 + t * 16 + l16;
        float* cp = Cm + (size_t)(m0 + sub * 8) * OUTC + ncol;
        #pragma unroll
        for (int v = 0; v < 8; ++v)
            cp[(size_t)v * OUTC] = acc[t][v];
    }
}

// ---------------------------------------------------------------------------
// edge aggregation: 4 edges per 256-thread block, 64 threads per edge
// msg = xw[etype, src, :]; atomic add into sums[(dst*R+etype)*256], cnt
// ---------------------------------------------------------------------------
__global__ void rgcn_edge_agg(const int* __restrict__ edge_index,
                              const int* __restrict__ edge_type,
                              const float* __restrict__ xw,
                              float* __restrict__ sums,
                              float* __restrict__ cnt,
                              int E, size_t xwStride) {
    int e = blockIdx.x * 4 + (threadIdx.x >> 6);
    if (e >= E) return;
    int tid = threadIdx.x & 63;
    int s = edge_index[e];
    int d = edge_index[E + e];
    int r = edge_type[e];
    const float4* m = (const float4*)(xw + (size_t)r * xwStride + (size_t)s * OUTC);
    float4 v = m[tid];
    float* dp = sums + ((size_t)d * RELS + r) * OUTC + tid * 4;
    atomicAdd(dp + 0, v.x);
    atomicAdd(dp + 1, v.y);
    atomicAdd(dp + 2, v.z);
    atomicAdd(dp + 3, v.w);
    if (tid == 0) atomicAdd(&cnt[(size_t)d * RELS + r], 1.0f);
}

// ---------------------------------------------------------------------------
// combine: h = sigmoid( sum_r sums/max(cnt,1) + root_xw + bias ); pad rows = 0
// ---------------------------------------------------------------------------
__global__ void rgcn_combine(const float* __restrict__ sums,
                             const float* __restrict__ cnt,
                             const float* __restrict__ rootxw,
                             const float* __restrict__ bias,
                             float* __restrict__ h32,
                             __bf16* __restrict__ hbf,
                             int Nn, int Npad) {
    size_t t = (size_t)blockIdx.x * 256 + threadIdx.x;
    size_t total = (size_t)Npad * OUTC;
    if (t >= total) return;
    int n = (int)(t / OUTC);
    int o = (int)(t % OUTC);
    float v = 0.f;
    if (n < Nn) {
        v = rootxw[(size_t)n * OUTC + o] + bias[o];
        #pragma unroll
        for (int r = 0; r < RELS; ++r) {
            float c = cnt[(size_t)n * RELS + r];
            v += sums[((size_t)n * RELS + r) * OUTC + o] / fmaxf(c, 1.0f);
        }
        v = 1.0f / (1.0f + __expf(-v));
    }
    h32[t] = v;
    hbf[t] = (__bf16)v;
}

// ---------------------------------------------------------------------------
// BN stats: grid (256 channels, node-chunks), LDS reduce then one atomic pair
// ---------------------------------------------------------------------------
__global__ void rgcn_bn_reduce(const float* __restrict__ g,
                               const float* __restrict__ g1_b,
                               float* __restrict__ bnsum,
                               float* __restrict__ bnsq, int Nn) {
    __shared__ float s1[256], s2[256];
    int c = blockIdx.x;
    int n = blockIdx.y * 256 + threadIdx.x;
    float v = 0.f;
    if (n < Nn) v = g[(size_t)n * OUTC + c] + g1_b[c];
    s1[threadIdx.x] = v;
    s2[threadIdx.x] = v * v;
    __syncthreads();
    for (int s = 128; s > 0; s >>= 1) {
        if (threadIdx.x < s) {
            s1[threadIdx.x] += s1[threadIdx.x + s];
            s2[threadIdx.x] += s2[threadIdx.x + s];
        }
        __syncthreads();
    }
    if (threadIdx.x == 0) {
        atomicAdd(&bnsum[c], s1[0]);
        atomicAdd(&bnsq[c],  s2[0]);
    }
}

// float ordered-uint encoding for atomic max
__device__ __forceinline__ unsigned f_ord(float f) {
    unsigned u = __float_as_uint(f);
    return (u & 0x80000000u) ? ~u : (u | 0x80000000u);
}
__device__ __forceinline__ float f_unord(unsigned u) {
    return (u & 0x80000000u) ? __uint_as_float(u & 0x7fffffffu) : __uint_as_float(~u);
}

// ---------------------------------------------------------------------------
// gate: normalize + relu + dot(g2_w); one wave per node; atomicMax per graph
// ---------------------------------------------------------------------------
__global__ void rgcn_gate(const float* __restrict__ g,
                          const float* __restrict__ g1_b,
                          const float* __restrict__ bnsum,
                          const float* __restrict__ bnsq,
                          const float* __restrict__ bn_g,
                          const float* __restrict__ bn_b,
                          const float* __restrict__ g2_w,
                          const float* __restrict__ g2_b,
                          const int* __restrict__ batch,
                          float* __restrict__ gate,
                          unsigned* __restrict__ gmaxu, int Nn) {
    int wave = threadIdx.x >> 5, lane = threadIdx.x & 31;
    int n = blockIdx.x * 8 + wave;
    if (n >= Nn) return;
    float invN = 1.0f / (float)Nn;
    float acc = 0.f;
    #pragma unroll
    for (int k = 0; k < 8; ++k) {
        int c = lane + k * 32;
        float v  = g[(size_t)n * OUTC + c] + g1_b[c];
        float mu = bnsum[c] * invN;
        float var = bnsq[c] * invN - mu * mu;
        v = (v - mu) * rsqrtf(var + 1e-5f) * bn_g[c] + bn_b[c];
        v = fmaxf(v, 0.f);
        acc += v * g2_w[c];
    }
    #pragma unroll
    for (int m = 16; m > 0; m >>= 1) acc += __shfl_xor(acc, m, 32);
    if (lane == 0) {
        float gt = acc + g2_b[0];
        gate[n] = gt;
        atomicMax(&gmaxu[batch[n]], f_ord(gt));
    }
}

__global__ void rgcn_exp(const float* __restrict__ gate,
                         const unsigned* __restrict__ gmaxu,
                         const int* __restrict__ batch,
                         float* __restrict__ ebuf,
                         float* __restrict__ denom, int Nn) {
    int n = blockIdx.x * 256 + threadIdx.x;
    if (n >= Nn) return;
    int b = batch[n];
    float e = __expf(gate[n] - f_unord(gmaxu[b]));
    ebuf[n] = e;
    atomicAdd(&denom[b], e);
}

__global__ void rgcn_pool(const float* __restrict__ ebuf,
                          const float* __restrict__ denom,
                          const float* __restrict__ h32,
                          const int* __restrict__ batch,
                          float* __restrict__ pooled, int Nn) {
    size_t t = (size_t)blockIdx.x * 256 + threadIdx.x;
    if (t >= (size_t)Nn * OUTC) return;
    int n = (int)(t / OUTC);
    int o = (int)(t % OUTC);
    int b = batch[n];
    float attn = ebuf[n] / (denom[b] + 1e-16f);
    atomicAdd(&pooled[(size_t)b * OUTC + o], attn * h32[(size_t)n * OUTC + o]);
}

__global__ void rgcn_final(const float* __restrict__ pooled,
                           const float* __restrict__ gl_w,
                           const float* __restrict__ gl_b,
                           float* __restrict__ out, int Bg) {
    int wave = threadIdx.x >> 5, lane = threadIdx.x & 31;
    int b = blockIdx.x * 8 + wave;
    if (b >= Bg) return;
    float acc = 0.f;
    #pragma unroll
    for (int k = 0; k < 8; ++k) {
        int c = lane + k * 32;
        acc += pooled[(size_t)b * OUTC + c] * gl_w[c];
    }
    #pragma unroll
    for (int m = 16; m > 0; m >>= 1) acc += __shfl_xor(acc, m, 32);
    if (lane == 0) out[b] = 1.0f / (1.0f + __expf(-(acc + gl_b[0])));
}

// ---------------------------------------------------------------------------
extern "C" void kernel_launch(void* const* d_in, const int* in_sizes, int n_in,
                              void* d_out, int out_size, void* d_ws, size_t ws_size,
                              hipStream_t stream) {
    const int*   x          = (const int*)d_in[0];
    const int*   edge_index = (const int*)d_in[1];
    const int*   edge_type  = (const int*)d_in[2];
    const int*   batch      = (const int*)d_in[3];
    const float* emb[6]     = {(const float*)d_in[4], (const float*)d_in[5],
                               (const float*)d_in[6], (const float*)d_in[7],
                               (const float*)d_in[8], (const float*)d_in[9]};
    const float* W1    = (const float*)d_in[10];
    const float* root1 = (const float*)d_in[11];
    const float* b1    = (const float*)d_in[12];
    const float* W2    = (const float*)d_in[13];
    const float* root2 = (const float*)d_in[14];
    const float* b2    = (const float*)d_in[15];
    const float* g1_w  = (const float*)d_in[16];
    const float* g1_b  = (const float*)d_in[17];
    const float* bn_g  = (const float*)d_in[18];
    const float* bn_b  = (const float*)d_in[19];
    const float* g2_w  = (const float*)d_in[20];
    const float* g2_b  = (const float*)d_in[21];
    const float* gl_w  = (const float*)d_in[22];
    const float* gl_b  = (const float*)d_in[23];
    float* out = (float*)d_out;

    const int Nn  = in_sizes[3];
    const int Ee  = in_sizes[2];
    const int Bg  = out_size;
    const int DIN = 384;
    const int Npad = ((Nn + 127) / 128) * 128;

    // ---- workspace carve-out (256B aligned) ----
    char* ws = (char*)d_ws;
    size_t off = 0;
    auto carve = [&](size_t bytes) { size_t o = off; off += (bytes + 255) & ~(size_t)255; return (void*)(ws + o); };

    __bf16* featsBf = (__bf16*)carve((size_t)Npad * DIN * 2);
    __bf16* wp1     = (__bf16*)carve((size_t)5 * DIN * OUTC * 2);
    __bf16* wp2     = (__bf16*)carve((size_t)5 * OUTC * OUTC * 2);
    __bf16* wpg     = (__bf16*)carve((size_t)OUTC * OUTC * 2);
    float*  xw      = (float*)carve((size_t)5 * Npad * OUTC * 4);   // also aliased as gbuf
    float*  sums    = (float*)carve((size_t)Nn * RELS * OUTC * 4);
    float*  cnt     = (float*)carve((size_t)Nn * RELS * 4);
    float*  h32     = (float*)carve((size_t)Npad * OUTC * 4);
    __bf16* hbf     = (__bf16*)carve((size_t)Npad * OUTC * 2);
    float*  bnsum   = (float*)carve(OUTC * 4);
    float*  bnsq    = (float*)carve(OUTC * 4);
    float*  gateb   = (float*)carve((size_t)Nn * 4);
    float*  ebuf    = (float*)carve((size_t)Nn * 4);
    unsigned* gmaxu = (unsigned*)carve((size_t)Bg * 4);
    float*  denom   = (float*)carve((size_t)Bg * 4);
    float*  pooled  = (float*)carve((size_t)Bg * OUTC * 4);
    float*  gbuf    = xw;   // gate GEMM output reuses xw (dead by then)

    auto fill = [&](void* p, unsigned v, size_t n_u32) {
        rgcn_fill_u32<<<2048, 256, 0, stream>>>((unsigned*)p, v, n_u32);
    };

    const size_t xwStride = (size_t)Npad * OUTC;

    // ---- pack weights ----
    for (int r = 0; r < RELS; ++r)
        rgcn_pack_w<<<(DIN * OUTC + 255) / 256, 256, 0, stream>>>(W1 + (size_t)r * DIN * OUTC, wp1 + (size_t)r * DIN * OUTC, DIN);
    rgcn_pack_w<<<(DIN * OUTC + 255) / 256, 256, 0, stream>>>(root1, wp1 + (size_t)4 * DIN * OUTC, DIN);
    for (int r = 0; r < RELS; ++r)
        rgcn_pack_w<<<(OUTC * OUTC + 255) / 256, 256, 0, stream>>>(W2 + (size_t)r * OUTC * OUTC, wp2 + (size_t)r * OUTC * OUTC, OUTC);
    rgcn_pack_w<<<(OUTC * OUTC + 255) / 256, 256, 0, stream>>>(root2, wp2 + (size_t)4 * OUTC * OUTC, OUTC);
    rgcn_pack_w<<<(OUTC * OUTC + 255) / 256, 256, 0, stream>>>(g1_w, wpg, OUTC);

    // ---- embedding gather ----
    {
        size_t total = (size_t)Npad * DIN;
        rgcn_gather<<<(unsigned)((total + 255) / 256), 256, 0, stream>>>(
            x, emb[0], emb[1], emb[2], emb[3], emb[4], emb[5], featsBf, Nn, Npad);
    }

    // ===== layer 1 =====
    rgcn_wmma_gemm<<<dim3(Npad / 128, 2, 5), 256, 0, stream>>>(
        featsBf, wp1, xw, DIN, (size_t)DIN * OUTC, xwStride);
    fill(sums, 0, (size_t)Nn * RELS * OUTC);
    fill(cnt, 0, (size_t)Nn * RELS);
    rgcn_edge_agg<<<(Ee + 3) / 4, 256, 0, stream>>>(edge_index, edge_type, xw, sums, cnt, Ee, xwStride);
    rgcn_combine<<<(unsigned)(((size_t)Npad * OUTC + 255) / 256), 256, 0, stream>>>(
        sums, cnt, xw + 4 * xwStride, b1, h32, hbf, Nn, Npad);

    // ===== layer 2 =====
    rgcn_wmma_gemm<<<dim3(Npad / 128, 2, 5), 256, 0, stream>>>(
        hbf, wp2, xw, OUTC, (size_t)OUTC * OUTC, xwStride);
    fill(sums, 0, (size_t)Nn * RELS * OUTC);
    fill(cnt, 0, (size_t)Nn * RELS);
    rgcn_edge_agg<<<(Ee + 3) / 4, 256, 0, stream>>>(edge_index, edge_type, xw, sums, cnt, Ee, xwStride);
    rgcn_combine<<<(unsigned)(((size_t)Npad * OUTC + 255) / 256), 256, 0, stream>>>(
        sums, cnt, xw + 4 * xwStride, b2, h32, hbf, Nn, Npad);

    // ===== gate GEMM (reuses xw as output) =====
    rgcn_wmma_gemm<<<dim3(Npad / 128, 2, 1), 256, 0, stream>>>(
        hbf, wpg, gbuf, OUTC, (size_t)OUTC * OUTC, xwStride);

    // ===== BN stats + gate + segment softmax + pool =====
    fill(bnsum, 0, OUTC);
    fill(bnsq, 0, OUTC);
    rgcn_bn_reduce<<<dim3(OUTC, (Nn + 255) / 256), 256, 0, stream>>>(gbuf, g1_b, bnsum, bnsq, Nn);

    fill(gmaxu, 0x00800000u /* ord(-FLT_MAX) */, Bg);
    fill(denom, 0, Bg);
    fill(pooled, 0, (size_t)Bg * OUTC);

    rgcn_gate<<<(Nn + 7) / 8, 256, 0, stream>>>(gbuf, g1_b, bnsum, bnsq, bn_g, bn_b,
                                                g2_w, g2_b, batch, gateb, gmaxu, Nn);
    rgcn_exp<<<(Nn + 255) / 256, 256, 0, stream>>>(gateb, gmaxu, batch, ebuf, denom, Nn);
    rgcn_pool<<<(unsigned)(((size_t)Nn * OUTC + 255) / 256), 256, 0, stream>>>(
        ebuf, denom, h32, batch, pooled, Nn);
    rgcn_final<<<(Bg + 7) / 8, 256, 0, stream>>>(pooled, gl_w, gl_b, out, Bg);
}
